// CausalMultiHeadSelfAttention_79027398246613
// MI455X (gfx1250) — matrix operand order repaired and checked by reading, not compiled, not run on hardware
//
#include <hip/hip_runtime.h>
#include <cstdint>

// ---------------------------------------------------------------------------
// Causal MHSA forward on gfx1250 (MI455X), bf16 WMMA path with fp32 accumulate
// + Tensor Data Mover (TDM) staging of K tiles in the attention kernel
// ---------------------------------------------------------------------------

typedef __attribute__((ext_vector_type(16))) __bf16 v16bf;
typedef __attribute__((ext_vector_type(8)))  __bf16 v8bf;
typedef __attribute__((ext_vector_type(8)))  float  v8f;
typedef __attribute__((ext_vector_type(4)))  unsigned ui32x4;
typedef __attribute__((ext_vector_type(8)))  int      i32x8;
typedef __attribute__((ext_vector_type(4)))  int      i32x4;

#define B_SZ   4
#define T_SZ   2048
#define DM     1024
#define H_SZ   16
#define DK     64
#define NQKV   3072
#define MROWS  (B_SZ * T_SZ)     // 8192

// round-to-nearest-even f32 -> bf16 (bit-level, no reliance on native cvt)
static __device__ __forceinline__ __bf16 f2bf(float f) {
  unsigned u = __builtin_bit_cast(unsigned, f);
  unsigned r = (u + 0x7FFFu + ((u >> 16) & 1u)) >> 16;
  unsigned short s = (unsigned short)r;
  return __builtin_bit_cast(__bf16, s);
}

static __device__ __forceinline__ v16bf pack16(v8bf lo, v8bf hi) {
  v16bf r;
#pragma unroll
  for (int i = 0; i < 8; ++i) { r[i] = lo[i]; r[i + 8] = hi[i]; }
  return r;
}

// ---------------------------------------------------------------------------
// TDM: DMA one 16x64 bf16 tile (row stride 64 elements) from global -> LDS
// D# packed per CDNA5 ISA 8.3/8.4 (2D tile, groups 2/3 zero).
// ---------------------------------------------------------------------------
static __device__ __forceinline__ void tdm_load_k_tile(const __bf16* g,
                                                       uint32_t lds_byte) {
  uint64_t ga = (uint64_t)(uintptr_t)g;
  ui32x4 g0;
  g0[0] = 1u;                                            // count=1, user mode
  g0[1] = lds_byte;                                      // lds_addr
  g0[2] = (uint32_t)ga;                                  // global_addr[31:0]
  g0[3] = (uint32_t)((ga >> 32) & 0x01FFFFFFu) | (2u << 30); // [56:32] | type=2
  i32x8 g1;
  g1[0] = (int)(1u << 16);       // workgroup_mask=0, data_size=1 (2 bytes)
  g1[1] = (int)(64u << 16);      // tensor_dim0 = 64 (low 16 bits)
  g1[2] = (int)(2048u << 16);    // tensor_dim0 hi=0 | tensor_dim1 = 2048 (lo16)
  g1[3] = (int)(64u << 16);      // tensor_dim1 hi=0 | tile_dim0 = 64
  g1[4] = 16;                    // tile_dim1 = 16, tile_dim2 = 0
  g1[5] = 64;                    // tensor_dim0_stride = 64 (low 32)
  g1[6] = 0;                     // stride0 hi | stride1 lo
  g1[7] = 0;                     // stride1 hi
  i32x4 z4 = {0, 0, 0, 0};
#if __clang_major__ >= 23
  i32x8 z8 = {0, 0, 0, 0, 0, 0, 0, 0};
  __builtin_amdgcn_tensor_load_to_lds(g0, g1, z4, z4, z8, 0);
#else
  __builtin_amdgcn_tensor_load_to_lds(g0, g1, z4, z4, 0);
#endif
}

// ---------------------------------------------------------------------------
// 1) element-wise fp32 -> bf16
// ---------------------------------------------------------------------------
__global__ void cvt_f32_bf16(const float* __restrict__ src,
                             __bf16* __restrict__ dst, int n) {
  int i = blockIdx.x * blockDim.x + threadIdx.x;
  if (i < n) dst[i] = f2bf(src[i]);
}

// 2) transpose-convert: W[K][N] fp32 -> Wt[N][K] bf16 (K-contiguous B operand)
__global__ void cvt_transpose_bf16(const float* __restrict__ W,
                                   __bf16* __restrict__ Wt, int K, int N) {
  int i = blockIdx.x * blockDim.x + threadIdx.x;
  if (i >= K * N) return;
  int k = i / N, n = i % N;
  Wt[(size_t)n * K + k] = f2bf(W[i]);
}

// ---------------------------------------------------------------------------
// 3) bf16 GEMM: C_f32[M][N] = A_bf[M][K] @ Bt_bf[N][K]^T
//    block = 256 threads = 8 waves (2x4); each wave computes a 32x64 tile
//    (2 M-frags x 4 N-frags register blocking -> 8 WMMA per 12 b128 loads)
//    block tile = 64(M) x 256(N)
// ---------------------------------------------------------------------------
__global__ __launch_bounds__(256) void gemm_bf16_tn(
    const __bf16* __restrict__ A, const __bf16* __restrict__ Bt,
    float* __restrict__ C, int M, int N, int K) {
  int tid  = threadIdx.x;
  int w    = tid >> 5;
  int lane = tid & 31;
  int h    = lane >> 4;     // half-wave
  int lm   = lane & 15;

  int m0 = blockIdx.y * 64  + (w & 1) * 32;   // wave: rows m0..m0+31
  int n0 = blockIdx.x * 256 + (w >> 1) * 64;  // wave: cols n0..n0+63

  const __bf16* Arow0 = A + (size_t)(m0 + lm) * K;
  const __bf16* Arow1 = A + (size_t)(m0 + 16 + lm) * K;
  const __bf16* Brow0 = Bt + (size_t)(n0 + lm) * K;

  v8f acc[2][4];
#pragma unroll
  for (int mi = 0; mi < 2; ++mi)
#pragma unroll
    for (int j = 0; j < 4; ++j) acc[mi][j] = (v8f){};

  for (int kb = 0; kb < K; kb += 32) {
    // A fragments (16x32): lane's two 8-elem runs at kb+h*8 and kb+16+h*8
    v16bf a0 = pack16(*(const v8bf*)(Arow0 + kb + h * 8),
                      *(const v8bf*)(Arow0 + kb + 16 + h * 8));
    v16bf a1 = pack16(*(const v8bf*)(Arow1 + kb + h * 8),
                      *(const v8bf*)(Arow1 + kb + 16 + h * 8));
#pragma unroll
    for (int j = 0; j < 4; ++j) {
      const __bf16* Br = Brow0 + (size_t)(j * 16) * K + kb + h * 8;
      v16bf b = pack16(*(const v8bf*)(Br), *(const v8bf*)(Br + 16));
      acc[0][j] = __builtin_amdgcn_wmma_f32_16x16x32_bf16(
          false, a0, false, b, (short)0, acc[0][j], false, false);
      acc[1][j] = __builtin_amdgcn_wmma_f32_16x16x32_bf16(
          false, a1, false, b, (short)0, acc[1][j], false, false);
    }
  }
#pragma unroll
  for (int mi = 0; mi < 2; ++mi)
#pragma unroll
    for (int j = 0; j < 4; ++j)
#pragma unroll
      for (int r = 0; r < 8; ++r) {
        int m = m0 + mi * 16 + r + 8 * h;
        C[(size_t)m * N + n0 + j * 16 + lm] = acc[mi][j][r];
      }
}

// ---------------------------------------------------------------------------
// 4) RoPE + split qkv_f32[M][3072] -> q_bf/k_bf [B,H,T,64], v transposed
//    vt_bf [B,H,64,T]. One thread per (b,t,h,pair).
// ---------------------------------------------------------------------------
__global__ void rope_split(const float* __restrict__ qkv,
                           const int* __restrict__ pos,
                           __bf16* __restrict__ qb, __bf16* __restrict__ kb,
                           __bf16* __restrict__ vt) {
  int tid = blockIdx.x * blockDim.x + threadIdx.x;   // B*T*H*32 threads
  int d2 = tid & 31;
  int h  = (tid >> 5) & 15;
  int t  = (tid >> 9) & (T_SZ - 1);
  int b  = tid >> 20;
  if (b >= B_SZ) return;

  size_t base = ((size_t)(b * T_SZ + t)) * NQKV + h * DK + 2 * d2;
  float q1 = qkv[base],        q2 = qkv[base + 1];
  float k1 = qkv[base + 1024], k2 = qkv[base + 1025];
  float v1 = qkv[base + 2048], v2 = qkv[base + 2049];

  float p  = (float)pos[t];
  float fr = __powf(10000.0f, -((float)(2 * d2) / 64.0f));
  float s, c;
  __sincosf(p * fr, &s, &c);

  size_t qk = (((size_t)(b * H_SZ + h)) * T_SZ + t) * DK + 2 * d2;
  qb[qk]     = f2bf(q1 * c - q2 * s);
  qb[qk + 1] = f2bf(q1 * s + q2 * c);
  kb[qk]     = f2bf(k1 * c - k2 * s);
  kb[qk + 1] = f2bf(k1 * s + k2 * c);

  size_t vti = (((size_t)(b * H_SZ + h)) * DK + 2 * d2) * T_SZ + t;
  vt[vti]        = f2bf(v1);
  vt[vti + T_SZ] = f2bf(v2);
}

// ---------------------------------------------------------------------------
// 5) attention: one workgroup (4 waves) per (b, h, 16-row query tile)
//    K tiles staged to LDS via TDM, double-buffered per wave.
// ---------------------------------------------------------------------------
#define SROW 2052      // fp32 S row stride (padded vs 2048 for banks)
#define PROW 2056      // bf16 P row stride
#define RED_OFF   (16 * SROW * 4 + 16 * PROW * 2)
#define STAGE_OFF (RED_OFF + 160 * 4)
#define SMEM_BYTES (STAGE_OFF + 4 * 2 * 16 * DK * 2)   // 4 waves x 2 bufs x 2KB

__global__ __launch_bounds__(128) void attn_kernel(
    const __bf16* __restrict__ qb, const __bf16* __restrict__ kbp,
    const __bf16* __restrict__ vt, __bf16* __restrict__ ctx) {
  extern __shared__ char smem[];
  float*  Sf    = (float*)smem;                     // [16][SROW]
  __bf16* P     = (__bf16*)(smem + 16 * SROW * 4);  // [16][PROW]
  float*  red   = (float*)(smem + RED_OFF);         // [16][8]
  float*  rmax  = red + 128;                        // [16]
  float*  rsum  = red + 144;                        // [16]
  __bf16* stage = (__bf16*)(smem + STAGE_OFF);      // 4 waves x 2 x 16x64

  int qt = blockIdx.x, hH = blockIdx.y, b = blockIdx.z;
  int bh = b * H_SZ + hH;
  int tid = threadIdx.x, wid = tid >> 5, lane = tid & 31;
  int hh = lane >> 4, lm = lane & 15;
  int q0 = qt * 16;
  int L = (qt + 1) * 16;
  int Lpad = (L + 31) & ~31;

  // 32-bit LDS byte offset of the staging region (for TDM descriptors)
  uint32_t stage_lds =
      (uint32_t)(size_t)(__attribute__((address_space(3))) char*)(void*)stage;

  // --- load Q fragments (16x64 = two 16x32 A fragments), reused for all K tiles
  const __bf16* Qrow = qb + ((size_t)bh * T_SZ + q0 + lm) * DK;
  v16bf aq0 = pack16(*(const v8bf*)(Qrow + hh * 8),
                     *(const v8bf*)(Qrow + 16 + hh * 8));
  v16bf aq1 = pack16(*(const v8bf*)(Qrow + 32 + hh * 8),
                     *(const v8bf*)(Qrow + 48 + hh * 8));

  // --- phase 1: S = (Q K^T) * 1/sqrt(64), causal-masked, into LDS
  const __bf16* Kbase = kbp + (size_t)bh * T_SZ * DK;
  if (wid <= qt)   // prime buffer 0
    tdm_load_k_tile(Kbase + (size_t)(wid * 16) * DK,
                    stage_lds + (wid * 2 + 0) * 2048);
  int i = 0;
  for (int kt = wid; kt <= qt; kt += 4, ++i) {
    int ktn = kt + 4;
    if (ktn <= qt) {
      tdm_load_k_tile(Kbase + (size_t)(ktn * 16) * DK,
                      stage_lds + (wid * 2 + ((i + 1) & 1)) * 2048);
      __builtin_amdgcn_s_wait_tensorcnt(1);   // in-order: tile i is resident
    } else {
      __builtin_amdgcn_s_wait_tensorcnt(0);
    }
    const __bf16* Kt = stage + (wid * 2 + (i & 1)) * (16 * DK) + lm * DK;
    v16bf bk0 = pack16(*(const v8bf*)(Kt + hh * 8),
                       *(const v8bf*)(Kt + 16 + hh * 8));
    v16bf bk1 = pack16(*(const v8bf*)(Kt + 32 + hh * 8),
                       *(const v8bf*)(Kt + 48 + hh * 8));
    v8f z = {};
    v8f acc = __builtin_amdgcn_wmma_f32_16x16x32_bf16(
        false, aq0, false, bk0, (short)0, z, false, false);
    acc = __builtin_amdgcn_wmma_f32_16x16x32_bf16(
        false, aq1, false, bk1, (short)0, acc, false, false);
#pragma unroll
    for (int j = 0; j < 8; ++j) {
      int m = j + 8 * hh;                 // local query row
      int kg = kt * 16 + lm;              // global key index
      float sv = (kg <= q0 + m) ? acc[j] * 0.125f : -1.0e30f;
      Sf[m * SROW + kt * 16 + lm] = sv;
    }
  }
  __syncthreads();

  // --- phase 2: row softmax (16 rows, 8 threads per row)
  int row = tid >> 3, sub = tid & 7;
  float mx = -3.0e38f;
  for (int c = sub; c < L; c += 8) mx = fmaxf(mx, Sf[row * SROW + c]);
  red[row * 8 + sub] = mx;
  __syncthreads();
  if (sub == 0) {
    float m = -3.0e38f;
#pragma unroll
    for (int k = 0; k < 8; ++k) m = fmaxf(m, red[row * 8 + k]);
    rmax[row] = m;
  }
  __syncthreads();
  float rm = rmax[row];
  float sm = 0.0f;
  for (int c = sub; c < L; c += 8) sm += __expf(Sf[row * SROW + c] - rm);
  __syncthreads();
  red[row * 8 + sub] = sm;
  __syncthreads();
  if (sub == 0) {
    float s = 0.0f;
#pragma unroll
    for (int k = 0; k < 8; ++k) s += red[row * 8 + k];
    rsum[row] = s;
  }
  __syncthreads();
  float inv = 1.0f / rsum[row];
  for (int c = sub; c < Lpad; c += 8) {
    float pv = (c < L) ? __expf(Sf[row * SROW + c] - rm) * inv : 0.0f;
    P[row * PROW + c] = f2bf(pv);
  }
  __syncthreads();

  // --- phase 3: ctx tile = P @ V, each wave owns one 16-wide Dk slice
  int c0 = wid * 16;
  const __bf16* Vrow = vt + ((size_t)bh * DK + c0 + lm) * T_SZ;
  v8f acc = {};
  int nch = Lpad >> 5;
  for (int c = 0; c < nch; ++c) {
    v16bf a = pack16(*(v8bf*)(P + lm * PROW + c * 32 + hh * 8),
                     *(v8bf*)(P + lm * PROW + c * 32 + 16 + hh * 8));
    v16bf bv = pack16(*(const v8bf*)(Vrow + c * 32 + hh * 8),
                      *(const v8bf*)(Vrow + c * 32 + 16 + hh * 8));
    acc = __builtin_amdgcn_wmma_f32_16x16x32_bf16(
        false, a, false, bv, (short)0, acc, false, false);
  }
#pragma unroll
  for (int j = 0; j < 8; ++j) {
    int m = j + 8 * hh;
    ctx[((size_t)(b * T_SZ) + q0 + m) * DM + hH * DK + c0 + lm] = f2bf(acc[j]);
  }
}

// ---------------------------------------------------------------------------
// launch
// ---------------------------------------------------------------------------
extern "C" void kernel_launch(void* const* d_in, const int* in_sizes, int n_in,
                              void* d_out, int out_size, void* d_ws,
                              size_t ws_size, hipStream_t stream) {
  (void)in_sizes; (void)n_in; (void)out_size; (void)ws_size;
  const float* x    = (const float*)d_in[0];
  const float* Wqkv = (const float*)d_in[1];
  const float* Wo   = (const float*)d_in[2];
  const int*   pos  = (const int*)d_in[3];
  float* out = (float*)d_out;

  char* ws = (char*)d_ws;
  size_t off = 0;
  __bf16* x_bf   = (__bf16*)(ws + off); off += (size_t)MROWS * DM * 2;
  __bf16* wqkv_t = (__bf16*)(ws + off); off += (size_t)NQKV * DM * 2;
  __bf16* wo_t   = (__bf16*)(ws + off); off += (size_t)DM * DM * 2;
  float*  qkv_f  = (float*)(ws + off);  off += (size_t)MROWS * NQKV * 4;
  __bf16* q_bf   = (__bf16*)(ws + off); off += (size_t)MROWS * DM * 2;
  __bf16* k_bf   = (__bf16*)(ws + off); off += (size_t)MROWS * DM * 2;
  __bf16* vt_bf  = (__bf16*)(ws + off); off += (size_t)MROWS * DM * 2;
  __bf16* ctx_bf = (__bf16*)(ws + off); off += (size_t)MROWS * DM * 2;

  // convert inputs to bf16 (weights transposed -> K-contiguous B operands)
  cvt_f32_bf16<<<(MROWS * DM) / 256, 256, 0, stream>>>(x, x_bf, MROWS * DM);
  cvt_transpose_bf16<<<(DM * NQKV) / 256, 256, 0, stream>>>(Wqkv, wqkv_t, DM, NQKV);
  cvt_transpose_bf16<<<(DM * DM) / 256, 256, 0, stream>>>(Wo, wo_t, DM, DM);

  // qkv = x @ W_qkv  (fp32 accumulate)
  gemm_bf16_tn<<<dim3(NQKV / 256, MROWS / 64), 256, 0, stream>>>(
      x_bf, wqkv_t, qkv_f, MROWS, NQKV, DM);

  // RoPE + split to q/k [B,H,T,64] and transposed v [B,H,64,T]
  rope_split<<<(B_SZ * T_SZ * H_SZ * 32) / 256, 256, 0, stream>>>(
      qkv_f, pos, q_bf, k_bf, vt_bf);

  // causal attention (TDM-staged K tiles)
  attn_kernel<<<dim3(T_SZ / 16, H_SZ, B_SZ), 128, SMEM_BYTES, stream>>>(
      q_bf, k_bf, vt_bf, ctx_bf);

  // out = ctx @ W_o
  gemm_bf16_tn<<<dim3(DM / 256, MROWS / 64), 256, 0, stream>>>(
      ctx_bf, wo_t, out, MROWS, DM, DM);
}
